// HybridTernaryLinear_43112881717531
// MI455X (gfx1250) — matrix operand order repaired and checked by reading, not compiled, not run on hardware
//
#include <hip/hip_runtime.h>
#include <hip/hip_bf16.h>

// ---------------------------------------------------------------------------
// Problem constants (from reference)
// ---------------------------------------------------------------------------
#define TOKENS 8192
#define IN_F   4096
#define OUT_F  4096
#define N_OUT  64
#define RANK   16
#define LORA_SCALING 2.0f

// GEMM tiling
#define BM 128
#define BN 128
#define BK 64

// ---------------------------------------------------------------------------
// Vector types
// ---------------------------------------------------------------------------
typedef float    v8f   __attribute__((ext_vector_type(8)));
typedef __bf16   v16bf __attribute__((ext_vector_type(16)));
typedef __bf16   v8bf  __attribute__((ext_vector_type(8)));
typedef __bf16   v4bf  __attribute__((ext_vector_type(4)));
typedef unsigned int u32x4 __attribute__((ext_vector_type(4)));
typedef int          i32x8 __attribute__((ext_vector_type(8)));
typedef int          i32x4 __attribute__((ext_vector_type(4)));

union Frag16 {
    v16bf v;
    v8bf  h[2];
};

// ---------------------------------------------------------------------------
// TDM: DMA a [rows=128 x cols=64] bf16 tile (row-major, row stride = `stride`
// elements) from global memory into LDS at byte offset `lds_off`.
// D# layout per CDNA5 ISA §8 (Tensor DMA Descriptor).
// This toolchain exposes the 6-arg builtin:
//   (u32x4 g0, i32x8 g1, i32x4 g2, i32x4 g3, i32x8 pad, i32 cpol)
// ---------------------------------------------------------------------------
static __device__ __forceinline__ void tdm_load_tile_bf16(
    const __bf16* gptr, unsigned lds_off, unsigned stride_elems,
    unsigned tensor_rows)
{
    unsigned long long ga = (unsigned long long)gptr;

    u32x4 g0;
    g0[0] = 1u;                                       // count=1, user desc
    g0[1] = lds_off;                                  // lds_addr (bytes)
    g0[2] = (unsigned)(ga & 0xFFFFFFFFu);             // global_addr[31:0]
    g0[3] = (unsigned)((ga >> 32) & 0x1FFFFFFu)       // global_addr[56:32]
          | (2u << 30);                               // type = 2 ("image")

    const unsigned td0 = stride_elems;                // tensor_dim0 (K extent)
    const unsigned td1 = tensor_rows;                 // tensor_dim1 (rows)
    i32x8 g1;
    g1[0] = (int)(1u << 16);                          // wg_mask=0, data_size=1 (2B)
    g1[1] = (int)((td0 & 0xFFFFu) << 16);             // barrier_addr=0 | dim0 lo16
    g1[2] = (int)(((td0 >> 16) & 0xFFFFu)             // dim0 hi16
                | ((td1 & 0xFFFFu) << 16));           // dim1 lo16
    g1[3] = (int)(((td1 >> 16) & 0xFFFFu)             // dim1 hi16
                | ((unsigned)BK << 16));              // tile_dim0 = 64
    g1[4] = (int)((unsigned)BM);                      // tile_dim1=128, tile_dim2=0
    g1[5] = (int)stride_elems;                        // tensor_dim0_stride lo32
    g1[6] = 0;                                        // stride0 hi16 | stride1 lo16
    g1[7] = 0;                                        // stride1 hi32

    i32x4 z4; z4[0]=0; z4[1]=0; z4[2]=0; z4[3]=0;     // groups 2/3 unused (2-D tile)
    i32x8 z8; z8[0]=0; z8[1]=0; z8[2]=0; z8[3]=0;
              z8[4]=0; z8[5]=0; z8[6]=0; z8[7]=0;

    __builtin_amdgcn_tensor_load_to_lds(g0, g1, z4, z4, z8, 0);
}

static __device__ __forceinline__ unsigned lds_offset_of(const void* p)
{
    // Low 32 bits of the generic (flat) address are the LDS byte offset
    // (shared aperture = {SHARED_BASE[15:0],16'b0} in addr[63:32]).
    return (unsigned)(unsigned long long)p;
}

// ---------------------------------------------------------------------------
// Kernel 1: x (f32) -> bf16
// ---------------------------------------------------------------------------
__global__ __launch_bounds__(256) void convert_x_bf16(
    const float* __restrict__ x, __bf16* __restrict__ xb)
{
    const long i = ((long)blockIdx.x * 256 + threadIdx.x) * 8;
    float4 a = *(const float4*)(x + i);
    float4 b = *(const float4*)(x + i + 4);
    v8bf o;
    o[0] = (__bf16)a.x; o[1] = (__bf16)a.y; o[2] = (__bf16)a.z; o[3] = (__bf16)a.w;
    o[4] = (__bf16)b.x; o[5] = (__bf16)b.y; o[6] = (__bf16)b.z; o[7] = (__bf16)b.w;
    *(v8bf*)(xb + i) = o;
}

// ---------------------------------------------------------------------------
// Kernel 2: W_eff = W_t + scatter(outliers) + LORA_SCALING * (B @ A), cast bf16.
// One block per output row; deterministic (no atomics).
// ---------------------------------------------------------------------------
__global__ __launch_bounds__(256) void build_weff(
    const float* __restrict__ Wt, const float* __restrict__ ow,
    const int* __restrict__ idx, const float* __restrict__ lA,
    const float* __restrict__ lB, __bf16* __restrict__ wb)
{
    const int o = blockIdx.x;
    __shared__ float sB[RANK];
    __shared__ int   sMatch[N_OUT];
    __shared__ int   sN;

    if (threadIdx.x == 0) {
        int c = 0;
        for (int j = 0; j < N_OUT; ++j)
            if (idx[j] == o) sMatch[c++] = j;
        sN = c;
    }
    if (threadIdx.x < RANK)
        sB[threadIdx.x] = LORA_SCALING * lB[(long)o * RANK + threadIdx.x];
    __syncthreads();

    const int nm = sN;
    for (int i4 = threadIdx.x; i4 < IN_F / 4; i4 += 256) {
        const long i = (long)i4 * 4;
        float4 acc = *(const float4*)(Wt + (long)o * IN_F + i);
        for (int m = 0; m < nm; ++m) {
            float4 v = *(const float4*)(ow + (long)sMatch[m] * IN_F + i);
            acc.x += v.x; acc.y += v.y; acc.z += v.z; acc.w += v.w;
        }
        #pragma unroll
        for (int r = 0; r < RANK; ++r) {
            const float s = sB[r];
            float4 v = *(const float4*)(lA + (long)r * IN_F + i);
            acc.x += s * v.x; acc.y += s * v.y; acc.z += s * v.z; acc.w += s * v.w;
        }
        v4bf q;
        q[0] = (__bf16)acc.x; q[1] = (__bf16)acc.y;
        q[2] = (__bf16)acc.z; q[3] = (__bf16)acc.w;
        *(v4bf*)(wb + (long)o * IN_F + i) = q;
    }
}

// ---------------------------------------------------------------------------
// Kernel 3: out[M,N] = Xb[M,K] @ Wb[N,K]^T + bias, bf16 WMMA, f32 accum.
// Block = 256 threads (8 waves), block tile 128x128, K-step 64,
// TDM double-buffered LDS staging.
// ---------------------------------------------------------------------------
__global__ __launch_bounds__(256, 2) void hybrid_gemm(
    const __bf16* __restrict__ Xb, const __bf16* __restrict__ Wb,
    const float* __restrict__ bias, float* __restrict__ out)
{
    __shared__ __bf16 sA[2][BM * BK];
    __shared__ __bf16 sW[2][BN * BK];

    const int tid   = threadIdx.x;
    const int wave  = tid >> 5;
    const int lane  = tid & 31;
    const int lhalf = lane >> 4;       // 0/1 (which 16-lane half)
    const int lrow  = lane & 15;
    const int waveM = wave >> 2;       // 0..1 -> 64 rows each
    const int waveN = wave & 3;        // 0..3 -> 32 cols each
    const long m0 = (long)blockIdx.y * BM;
    const long n0 = (long)blockIdx.x * BN;

    v8f acc[4][2] = {};

    if (wave == 0) {
        tdm_load_tile_bf16(Xb + m0 * IN_F, lds_offset_of(&sA[0][0]), IN_F, TOKENS);
        tdm_load_tile_bf16(Wb + n0 * IN_F, lds_offset_of(&sW[0][0]), IN_F, OUT_F);
    }

    const int NKS = IN_F / BK;         // 64 K-steps
    for (int ks = 0; ks < NKS; ++ks) {
        const int cur = ks & 1;
        if (wave == 0) {
            if (ks + 1 < NKS) {
                const int  nxt = cur ^ 1;
                const long k   = (long)(ks + 1) * BK;
                tdm_load_tile_bf16(Xb + m0 * IN_F + k, lds_offset_of(&sA[nxt][0]), IN_F, TOKENS);
                tdm_load_tile_bf16(Wb + n0 * IN_F + k, lds_offset_of(&sW[nxt][0]), IN_F, OUT_F);
                __builtin_amdgcn_s_wait_tensorcnt(2);  // current buffer's 2 ops done
            } else {
                __builtin_amdgcn_s_wait_tensorcnt(0);
            }
        }
        __syncthreads();

        const __bf16* tA = &sA[cur][0];
        const __bf16* tW = &sW[cur][0];

        #pragma unroll
        for (int kk = 0; kk < BK; kk += 32) {
            // B fragments (32x16, K x N): lane half 0 holds K {0-7,16-23}, half 1 {8-15,24-31}
            Frag16 bfr[2];
            #pragma unroll
            for (int nt = 0; nt < 2; ++nt) {
                const __bf16* p = tW + (waveN * 32 + nt * 16 + lrow) * BK + kk + lhalf * 8;
                bfr[nt].h[0] = *(const v8bf*)(p);
                bfr[nt].h[1] = *(const v8bf*)(p + 16);
            }
            #pragma unroll
            for (int mt = 0; mt < 4; ++mt) {
                // A fragment (16x32, M x K): lane half 0 holds K {0-7,16-23}, half 1 {8-15,24-31}
                Frag16 afr;
                const __bf16* p = tA + (waveM * 64 + mt * 16 + lrow) * BK + kk + lhalf * 8;
                afr.h[0] = *(const v8bf*)(p);
                afr.h[1] = *(const v8bf*)(p + 16);
                #pragma unroll
                for (int nt = 0; nt < 2; ++nt) {
                    acc[mt][nt] = __builtin_amdgcn_wmma_f32_16x16x32_bf16(
                        false, afr.v, false, bfr[nt].v,
                        (short)0, acc[mt][nt], false, false);
                }
            }
        }
        __syncthreads();
    }

    // Epilogue: C layout -> M = r + lhalf*8, N = lrow within each 16x16 tile.
    #pragma unroll
    for (int mt = 0; mt < 4; ++mt) {
        #pragma unroll
        for (int nt = 0; nt < 2; ++nt) {
            const long col  = n0 + waveN * 32 + nt * 16 + lrow;
            const long rowb = m0 + waveM * 64 + mt * 16 + lhalf * 8;
            const float bv  = bias[col];
            #pragma unroll
            for (int r = 0; r < 8; ++r)
                out[(rowb + r) * OUT_F + col] = acc[mt][nt][r] + bv;
        }
    }
}

// ---------------------------------------------------------------------------
// Launch
// ---------------------------------------------------------------------------
extern "C" void kernel_launch(void* const* d_in, const int* in_sizes, int n_in,
                              void* d_out, int out_size, void* d_ws, size_t ws_size,
                              hipStream_t stream)
{
    const float* x    = (const float*)d_in[0];
    const float* Wt   = (const float*)d_in[1];
    const float* ow   = (const float*)d_in[2];
    const int*   idx  = (const int*)d_in[3];
    const float* lA   = (const float*)d_in[4];
    const float* lB   = (const float*)d_in[5];
    const float* bias = (const float*)d_in[6];
    float* out = (float*)d_out;

    __bf16* xb = (__bf16*)d_ws;                                       // 64 MB
    __bf16* wb = (__bf16*)((char*)d_ws + (size_t)TOKENS * IN_F * 2);  // +32 MB

    const long xelems = (long)TOKENS * IN_F;
    convert_x_bf16<<<(unsigned)(xelems / (256 * 8)), 256, 0, stream>>>(x, xb);
    build_weff<<<OUT_F, 256, 0, stream>>>(Wt, ow, idx, lA, lB, wb);

    dim3 grid(OUT_F / BN, TOKENS / BM);
    hybrid_gemm<<<grid, 256, 0, stream>>>(xb, wb, bias, out);
}
